// DeepAR_11338713661931
// MI455X (gfx1250) — matrix operand order repaired and checked by reading, not compiled, not run on hardware
//
#include <hip/hip_runtime.h>
#include <hip/hip_bf16.h>
#include <stdint.h>

// ---------------- problem constants ----------------
#define B_   512
#define SEQ_ 168
#define H_   24
#define F_   32
#define E_   64
#define HID_ 1024
#define TOT_ (SEQ_ + H_)          // 192
#define KX_  (F_ + E_)            // 96 = 3 K-chunks of 32
#define K0_  (KX_ + HID_)         // 1120
#define K1_  (2 * HID_)           // 2048
#define N_   (4 * HID_)           // 4096

typedef __attribute__((ext_vector_type(16))) __bf16 v16bf;
typedef __attribute__((ext_vector_type(8)))  float  v8f;

union FragU { uint4 q[2]; v16bf v; };

// float -> bf16 (round to nearest even)
__device__ __forceinline__ unsigned short f2bf(float f) {
    union { float f; uint32_t u; } v; v.f = f;
    uint32_t u = v.u;
    if ((u & 0x7fffffffu) > 0x7f800000u) return (unsigned short)((u >> 16) | 0x40);
    return (unsigned short)((u + 0x7fffu + ((u >> 16) & 1u)) >> 16);
}

__device__ __forceinline__ float sigm(float x) { return 1.0f / (1.0f + expf(-x)); }

// ---------------- once-per-launch kernels ----------------
__global__ void zero_f32_kernel(float* p, size_t n) {
    size_t i = (size_t)blockIdx.x * blockDim.x + threadIdx.x;
    if (i < n) p[i] = 0.0f;
}
__global__ void zero_u16_kernel(unsigned short* p, size_t n) {
    size_t i = (size_t)blockIdx.x * blockDim.x + threadIdx.x;
    if (i < n) p[i] = 0;
}
__global__ void pack_bias_kernel(const float* __restrict__ ba, const float* __restrict__ bb,
                                 float* __restrict__ o, int n) {
    int i = blockIdx.x * blockDim.x + threadIdx.x;
    if (i < n) o[i] = ba[i] + bb[i];
}

// Pack [Wih | Whh] into WMMA B-fragment order: blocks of (kc 32)x(nt 16), 512 halves
// per block; lane L owns 16 contiguous halves: n = nt*16 + (L&15), k = kc*32 + ((L>>4)<<4) + i
__global__ void pack_w_kernel(const float* __restrict__ Wih, const float* __restrict__ Whh,
                              unsigned short* __restrict__ Wp, int Kin, int Ktot, int N) {
    size_t pid = (size_t)blockIdx.x * blockDim.x + threadIdx.x;
    size_t total = (size_t)Ktot * N;
    if (pid >= total) return;
    int    r      = (int)(pid & 511);
    size_t blocki = pid >> 9;
    int lane = r >> 4, i = r & 15;
    int ntiles = N >> 4;
    int kcIdx = (int)(blocki / ntiles);
    int nt    = (int)(blocki % ntiles);
    int n = nt * 16 + (lane & 15);
    int k = kcIdx * 32 + ((lane >> 4) << 4) + i;
    float v = (k < Kin) ? Wih[(size_t)n * Kin + k]
                        : Whh[(size_t)n * (Ktot - Kin) + (k - Kin)];
    Wp[pid] = f2bf(v);
}

// Precompute XE[t][b][96] = x_t ∥ y-embed for ALL steps (teacher-forced embeds for
// t < SEQ; decode-phase embed slots zeroed, filled at runtime by cell1_head of t-1).
__global__ void build_xe_kernel(const float* __restrict__ X, const float* __restrict__ y,
                                const float* __restrict__ Xf, const float* __restrict__ Wemb,
                                const float* __restrict__ bemb, unsigned short* __restrict__ XE) {
    size_t idx = (size_t)blockIdx.x * blockDim.x + threadIdx.x;
    if (idx >= (size_t)TOT_ * B_ * KX_) return;
    int col = (int)(idx % KX_);
    size_t tb = idx / KX_;
    int b = (int)(tb % B_);
    int t = (int)(tb / B_);
    float v;
    if (col < F_) {
        v = (t < SEQ_) ? X[((size_t)b * SEQ_ + t) * F_ + col]
                       : Xf[((size_t)b * H_ + (t - SEQ_)) * F_ + col];
    } else if (t < SEQ_) {
        int e = col - F_;
        v = y[(size_t)b * SEQ_ + t] * Wemb[e] + bemb[e];
    } else {
        v = 0.0f;   // overwritten by cell1_head of step t-1 before use
    }
    XE[idx] = f2bf(v);
}

// ---------------- per-step kernels ----------------
// gates[M,N] = [Ax | Ah] @ Bp + bias, split A at K = ks (multiple of 32).
// One wave -> 16x128 strip; K loop double-buffered (A frag + 8 B frags in flight).
__global__ __launch_bounds__(256, 1) void wmma_gemm_kernel(
        const unsigned short* __restrict__ Ax, int ldx, int ks,
        const unsigned short* __restrict__ Ah, int ldh,
        const unsigned short* __restrict__ Bp, const float* __restrict__ bias,
        float* __restrict__ C, int M, int N, int K) {
    const int lane = threadIdx.x & 31;
    const int wave = threadIdx.x >> 5;
    const int gw   = blockIdx.x * 8 + wave;
    const int mTiles = M >> 4;
    const int rowTile  = gw % mTiles;
    const int colGroup = gw / mTiles;          // each covers 128 columns
    const int n0 = colGroup * 128;
    const int ntiles = N >> 4;
    const int nb0 = n0 >> 4;

    const int m = rowTile * 16 + (lane & 15);
    const int aKoff = (lane >> 4) * 16;                // lanes 16-31 start at K=16 within chunk
    const unsigned short* aRow0 = Ax + (size_t)m * ldx;
    const unsigned short* aRow1 = Ah + (size_t)m * ldh - ks;   // index with absolute kc

    v8f acc[8];
#pragma unroll
    for (int i = 0; i < 8; ++i) acc[i] = (v8f){0.f,0.f,0.f,0.f,0.f,0.f,0.f,0.f};

    auto loadA = [&](int kc) -> FragU {
        FragU a;
        const unsigned short* r = (kc < ks) ? aRow0 : aRow1;
        a.q[0] = *(const uint4*)(r + kc + aKoff);        // K 0-7 / 16-23
        a.q[1] = *(const uint4*)(r + kc + aKoff + 8);    // K 8-15 / 24-31
        return a;
    };
    auto bBaseOf = [&](int kc) -> const unsigned short* {
        return Bp + ((size_t)(kc >> 5) * ntiles + nb0) * 512 + lane * 16;
    };

    FragU a_cur = loadA(0);
    FragU b_cur[8];
    {
        const unsigned short* bb = bBaseOf(0);
#pragma unroll
        for (int nt = 0; nt < 8; ++nt) {
            b_cur[nt].q[0] = *(const uint4*)(bb + nt * 512);
            b_cur[nt].q[1] = *(const uint4*)(bb + nt * 512 + 8);
        }
    }

    int kc = 0;
    for (; kc + 32 < K; kc += 32) {
        // issue next chunk's loads before consuming the current fragments
        FragU a_nxt = loadA(kc + 32);
        FragU b_nxt[8];
        const unsigned short* bb = bBaseOf(kc + 32);
        __builtin_prefetch(bb + (size_t)ntiles * 512, 0, 0);   // K-chunk after next (L2)
#pragma unroll
        for (int nt = 0; nt < 8; ++nt) {
            b_nxt[nt].q[0] = *(const uint4*)(bb + nt * 512);
            b_nxt[nt].q[1] = *(const uint4*)(bb + nt * 512 + 8);
        }
#pragma unroll
        for (int nt = 0; nt < 8; ++nt)
            acc[nt] = __builtin_amdgcn_wmma_f32_16x16x32_bf16(
                false, a_cur.v, false, b_cur[nt].v, (short)0, acc[nt], false, false);
        a_cur = a_nxt;
#pragma unroll
        for (int nt = 0; nt < 8; ++nt) b_cur[nt] = b_nxt[nt];
    }
    // peeled last chunk
#pragma unroll
    for (int nt = 0; nt < 8; ++nt)
        acc[nt] = __builtin_amdgcn_wmma_f32_16x16x32_bf16(
            false, a_cur.v, false, b_cur[nt].v, (short)0, acc[nt], false, false);

    // C/D layout: VGPR r -> M = r (lanes 0-15), r+8 (lanes 16-31); N = lane & 15
    const int rBase = rowTile * 16 + ((lane >> 4) << 3);
    const int cBase = n0 + (lane & 15);
#pragma unroll
    for (int nt = 0; nt < 8; ++nt) {
        int col = cBase + nt * 16;
        float bv = bias[col];
#pragma unroll
        for (int r = 0; r < 8; ++r)
            C[(size_t)(rBase + r) * N + col] = acc[nt][r] + bv;
    }
}

// LSTM cell 0 pointwise; writes h0 (bf16) into H0
__global__ void cell0_kernel(const float* __restrict__ gates, float* __restrict__ c0,
                             unsigned short* __restrict__ H0) {
    int idx = blockIdx.x * blockDim.x + threadIdx.x;
    if (idx >= B_ * HID_) return;
    int b = idx >> 10, j = idx & (HID_ - 1);
    const float* g = gates + (size_t)b * N_;
    float ig = sigm(g[j]);
    float fg = sigm(g[HID_ + j]);
    float gg = tanhf(g[2 * HID_ + j]);
    float og = sigm(g[3 * HID_ + j]);
    float c = fg * c0[idx] + ig * gg;
    c0[idx] = c;
    H0[idx] = f2bf(og * tanhf(c));
}

// LSTM cell 1 pointwise + Gaussian heads + sampling + next-step y-embed.
// One block (1024 threads = 32 waves) per batch row.
__global__ __launch_bounds__(1024) void cell1_head_kernel(
        const float* __restrict__ gates, float* __restrict__ c1,
        unsigned short* __restrict__ H1, const float* __restrict__ Wmu,
        const float* __restrict__ bmu, const float* __restrict__ Wsig,
        const float* __restrict__ bsig, const float* __restrict__ eps,
        const float* __restrict__ Wemb, const float* __restrict__ bemb,
        unsigned short* __restrict__ XE, float* __restrict__ out, int t) {
    __shared__ float redm[32];
    __shared__ float reds[32];
    __shared__ float sy;
    int b = blockIdx.x;
    int j = threadIdx.x;
    const float* g = gates + (size_t)b * N_;
    float ig = sigm(g[j]);
    float fg = sigm(g[HID_ + j]);
    float gg = tanhf(g[2 * HID_ + j]);
    float og = sigm(g[3 * HID_ + j]);
    size_t idx = (size_t)b * HID_ + j;
    float c = fg * c1[idx] + ig * gg;
    c1[idx] = c;
    float h = og * tanhf(c);
    H1[idx] = f2bf(h);
    float hs = h > 0.f ? h : 0.f;
    float pm = hs * Wmu[j];
    float ps = hs * Wsig[j];
    // wave32 shuffle reduction, then one cross-wave pass
#pragma unroll
    for (int s = 16; s > 0; s >>= 1) {
        pm += __shfl_down(pm, s, 32);
        ps += __shfl_down(ps, s, 32);
    }
    int wid = j >> 5, ln = j & 31;
    if (ln == 0) { redm[wid] = pm; reds[wid] = ps; }
    __syncthreads();
    if (wid == 0) {
        float m  = redm[ln];
        float sg = reds[ln];
#pragma unroll
        for (int s = 16; s > 0; s >>= 1) {
            m  += __shfl_down(m, s, 32);
            sg += __shfl_down(sg, s, 32);
        }
        if (ln == 0) {
            float mu = m + bmu[0];
            float sr = sg + bsig[0];
            float sigma = ((sr > 20.f) ? sr : log1pf(expf(sr))) + 1e-6f;
            float ys = mu + sigma * eps[(size_t)t * B_ + b];
            float* mu_out = out + (size_t)B_ * H_;
            float* sg_out = mu_out + (size_t)B_ * TOT_;
            mu_out[(size_t)b * TOT_ + t] = mu;
            sg_out[(size_t)b * TOT_ + t] = sigma;
            if (t >= SEQ_ - 1 && t < SEQ_ - 1 + H_)
                out[(size_t)b * H_ + (t - (SEQ_ - 1))] = ys;
            sy = ys;
        }
    }
    __syncthreads();
    // fused "prep" for step t+1: decode-phase y-embed from the fresh sample
    if (t + 1 >= SEQ_ && t + 1 < TOT_ && j < E_) {
        XE[((size_t)(t + 1) * B_ + b) * KX_ + F_ + j] = f2bf(sy * Wemb[j] + bemb[j]);
    }
}

// ---------------- launcher ----------------
extern "C" void kernel_launch(void* const* d_in, const int* in_sizes, int n_in,
                              void* d_out, int out_size, void* d_ws, size_t ws_size,
                              hipStream_t stream) {
    const float* X     = (const float*)d_in[0];
    const float* y     = (const float*)d_in[1];
    const float* Xf    = (const float*)d_in[2];
    const float* Wemb  = (const float*)d_in[3];
    const float* bemb  = (const float*)d_in[4];
    const float* Wih0  = (const float*)d_in[5];
    const float* Whh0  = (const float*)d_in[6];
    const float* bih0  = (const float*)d_in[7];
    const float* bhh0  = (const float*)d_in[8];
    const float* Wih1  = (const float*)d_in[9];
    const float* Whh1  = (const float*)d_in[10];
    const float* bih1  = (const float*)d_in[11];
    const float* bhh1  = (const float*)d_in[12];
    const float* Wmu   = (const float*)d_in[13];
    const float* bmu   = (const float*)d_in[14];
    const float* Wsig  = (const float*)d_in[15];
    const float* bsig  = (const float*)d_in[16];
    const float* eps   = (const float*)d_in[17];
    float* out = (float*)d_out;

    char* ws = (char*)d_ws;
    size_t off = 0;
    auto alloc = [&](size_t bytes) -> void* {
        void* p = ws + off;
        off = (off + bytes + 255) & ~(size_t)255;
        return p;
    };
    unsigned short* W0p = (unsigned short*)alloc((size_t)K0_ * N_ * 2);
    unsigned short* W1p = (unsigned short*)alloc((size_t)K1_ * N_ * 2);
    float*          b0  = (float*)alloc((size_t)N_ * 4);
    float*          b1  = (float*)alloc((size_t)N_ * 4);
    unsigned short* XE  = (unsigned short*)alloc((size_t)TOT_ * B_ * KX_ * 2);
    unsigned short* H0  = (unsigned short*)alloc((size_t)B_ * HID_ * 2);
    unsigned short* H1  = (unsigned short*)alloc((size_t)B_ * HID_ * 2);
    float*          gts = (float*)alloc((size_t)B_ * N_ * 4);
    float*          c0  = (float*)alloc((size_t)B_ * HID_ * 4);
    float*          c1  = (float*)alloc((size_t)B_ * HID_ * 4);
    (void)ws_size; (void)in_sizes; (void)n_in; (void)out_size;

    // --- once per launch: pack weights/biases, precompute XE, zero state ---
    {
        size_t t0 = (size_t)K0_ * N_;
        pack_w_kernel<<<(unsigned)((t0 + 255) / 256), 256, 0, stream>>>(
            Wih0, Whh0, W0p, KX_, K0_, N_);
        size_t t1 = (size_t)K1_ * N_;
        pack_w_kernel<<<(unsigned)((t1 + 255) / 256), 256, 0, stream>>>(
            Wih1, Whh1, W1p, HID_, K1_, N_);
        pack_bias_kernel<<<(N_ + 255) / 256, 256, 0, stream>>>(bih0, bhh0, b0, N_);
        pack_bias_kernel<<<(N_ + 255) / 256, 256, 0, stream>>>(bih1, bhh1, b1, N_);
        size_t nxe = (size_t)TOT_ * B_ * KX_;
        build_xe_kernel<<<(unsigned)((nxe + 255) / 256), 256, 0, stream>>>(
            X, y, Xf, Wemb, bemb, XE);
        size_t nf = (size_t)B_ * HID_;
        zero_f32_kernel<<<(unsigned)((nf + 255) / 256), 256, 0, stream>>>(c0, nf);
        zero_f32_kernel<<<(unsigned)((nf + 255) / 256), 256, 0, stream>>>(c1, nf);
        zero_u16_kernel<<<(unsigned)((nf + 255) / 256), 256, 0, stream>>>(H0, nf);
        zero_u16_kernel<<<(unsigned)((nf + 255) / 256), 256, 0, stream>>>(H1, nf);
    }

    const int gemmBlocks = ((B_ / 16) * (N_ / 128)) / 8;   // 128 blocks x 8 waves

    for (int t = 0; t < TOT_; ++t) {
        const unsigned short* XEt = XE + (size_t)t * B_ * KX_;
        wmma_gemm_kernel<<<gemmBlocks, 256, 0, stream>>>(
            XEt, KX_, KX_, H0, HID_, W0p, b0, gts, B_, N_, K0_);
        cell0_kernel<<<(B_ * HID_) / 256, 256, 0, stream>>>(gts, c0, H0);
        wmma_gemm_kernel<<<gemmBlocks, 256, 0, stream>>>(
            H0, HID_, HID_, H1, HID_, W1p, b1, gts, B_, N_, K1_);
        cell1_head_kernel<<<B_, HID_, 0, stream>>>(
            gts, c1, H1, Wmu, bmu, Wsig, bsig, eps, Wemb, bemb, XE, out, t);
    }
}
